// RecommendationModel_4260607558138
// MI455X (gfx1250) — compile-verified, matrix operand order checked by reading, not run on hardware
//
#include <hip/hip_runtime.h>
#include <hip/hip_bf16.h>

// ---------------------------------------------------------------------------
// Hetero-GraphSAGE forward for MI455X (gfx1250, wave32, WMMA bf16 16x16x32)
// ---------------------------------------------------------------------------

typedef __attribute__((ext_vector_type(16))) __bf16 v16bf;
typedef __attribute__((ext_vector_type(8)))  float  v8f;

#define Hdim 64
#define NUc 200000
#define NMc 100000
#define NGc 32
#define NDc 20000
#define E1c 1000000
#define E2c 300000
#define E3c 100000
#define ELc 500000

// K index held by (vgpr v, halfword j) for a lane in half `half` (0/1),
// per CDNA5 ISA 16-bit A/B 16x32 layout.
static __device__ __forceinline__ int kmap(int v, int j, int half) {
  return ((v & 4) << 2) + half * 8 + ((v & 3) << 1) + j;
}

// Pack two contiguous 8-float spans (per-lane WMMA A/B footprint) into v16bf.
static __device__ __forceinline__ v16bf pack16(const float* __restrict__ s0) {
  const float4 f0 = *(const float4*)(s0);
  const float4 f1 = *(const float4*)(s0 + 4);
  const float4 f2 = *(const float4*)(s0 + 16);
  const float4 f3 = *(const float4*)(s0 + 20);
  v16bf a;
  a[0]  = (__bf16)f0.x; a[1]  = (__bf16)f0.y; a[2]  = (__bf16)f0.z; a[3]  = (__bf16)f0.w;
  a[4]  = (__bf16)f1.x; a[5]  = (__bf16)f1.y; a[6]  = (__bf16)f1.z; a[7]  = (__bf16)f1.w;
  a[8]  = (__bf16)f2.x; a[9]  = (__bf16)f2.y; a[10] = (__bf16)f2.z; a[11] = (__bf16)f2.w;
  a[12] = (__bf16)f3.x; a[13] = (__bf16)f3.y; a[14] = (__bf16)f3.z; a[15] = (__bf16)f3.w;
  return a;
}

// Same but scaled (folds the segment-mean divide into the A operand).
static __device__ __forceinline__ v16bf pack16s(const float* __restrict__ s0,
                                                float sc) {
  const float4 f0 = *(const float4*)(s0);
  const float4 f1 = *(const float4*)(s0 + 4);
  const float4 f2 = *(const float4*)(s0 + 16);
  const float4 f3 = *(const float4*)(s0 + 20);
  v16bf a;
  a[0]  = (__bf16)(f0.x * sc); a[1]  = (__bf16)(f0.y * sc);
  a[2]  = (__bf16)(f0.z * sc); a[3]  = (__bf16)(f0.w * sc);
  a[4]  = (__bf16)(f1.x * sc); a[5]  = (__bf16)(f1.y * sc);
  a[6]  = (__bf16)(f1.z * sc); a[7]  = (__bf16)(f1.w * sc);
  a[8]  = (__bf16)(f2.x * sc); a[9]  = (__bf16)(f2.y * sc);
  a[10] = (__bf16)(f2.z * sc); a[11] = (__bf16)(f2.w * sc);
  a[12] = (__bf16)(f3.x * sc); a[13] = (__bf16)(f3.y * sc);
  a[14] = (__bf16)(f3.z * sc); a[15] = (__bf16)(f3.w * sc);
  return a;
}

// ---------------------------------------------------------------------------
// Swizzle B[K x 64] (f32, row-major) into per-lane WMMA operand order (bf16):
// out[(((kc*4)+tn)*32 + lane)*16 + (2v+j)] = B[kc*32+kmap(v,j,half)][tn*16+l15]
// ---------------------------------------------------------------------------
__global__ __launch_bounds__(256) void k_swizzleB(
    __bf16* __restrict__ out, const float* __restrict__ B, int n /* K*64 */) {
  const int i = blockIdx.x * blockDim.x + threadIdx.x;
  if (i >= n) return;
  const int e    = i & 15;
  const int lane = (i >> 4) & 31;
  const int tn   = (i >> 9) & 3;
  const int kc   = i >> 11;
  const int kk   = kc * 32 + kmap(e >> 1, e & 1, lane >> 4);
  out[i] = (__bf16)B[(size_t)kk * Hdim + tn * 16 + (lane & 15)];
}

// ---------------------------------------------------------------------------
// Encode path: C[M x 64] += A[M x K] @ B[K x 64] (B pre-swizzled bf16).
// ---------------------------------------------------------------------------
template <int K, int TILES>
__global__ __launch_bounds__(256) void k_wmma_gemm_accum(
    float* __restrict__ C, const float* __restrict__ A,
    const __bf16* __restrict__ Bsw, int M) {
  constexpr int KC = K / 32;
  const int lane = threadIdx.x & 31;
  const int wave = threadIdx.x >> 5;
  const int tn   = wave & 3;
  const int half = lane >> 4;
  const int l15  = lane & 15;
  const int col  = tn * 16 + l15;

  v16bf b[KC];
#pragma unroll
  for (int kc = 0; kc < KC; ++kc)
    b[kc] = *(const v16bf*)(Bsw + ((size_t)(kc * 4 + tn) * 32 + lane) * 16);

  const int tm0 = blockIdx.x * (2 * TILES) + (wave >> 2) * TILES;
#pragma unroll
  for (int t = 0; t < TILES; ++t) {
    const int tm = tm0 + t;
    if (tm * 16 < M) {  // wave-uniform: EXEC untouched, WMMA stays legal
      const float* Arow = A + (size_t)(tm * 16 + l15) * K;
      v8f c = {0.f, 0.f, 0.f, 0.f, 0.f, 0.f, 0.f, 0.f};
#pragma unroll
      for (int kc = 0; kc < KC; ++kc) {
        const v16bf a = pack16(Arow + kc * 32 + half * 8);
        c = __builtin_amdgcn_wmma_f32_16x16x32_bf16(false, a, false, b[kc],
                                                    (short)0, c, false, false);
      }
      float* Crow = C + (size_t)(tm * 16) * Hdim + col;
#pragma unroll
      for (int r = 0; r < 8; ++r)
        Crow[(size_t)(r + 8 * half) * Hdim] += c[r];
    }
  }
}

// ---------------------------------------------------------------------------
// Fused conv output:  C = relu( bias + sum_t (A_t * (1/max(cnt_t,1))) @ B_t )
// NT terms (term 0 = x@sum(Wr), cnt=null; terms 1.. = agg@Wl with mean fold).
// All NT*2 B operands live in VGPRs; C is write-only. M % 16 == 0.
// ---------------------------------------------------------------------------
template <int NT, int TILES>
__global__ __launch_bounds__(256) void k_fused_out(
    float* __restrict__ C,
    const float* __restrict__ A0, const float* __restrict__ cnt0,
    const float* __restrict__ A1, const float* __restrict__ cnt1,
    const float* __restrict__ A2, const float* __restrict__ cnt2,
    const float* __restrict__ A3, const float* __restrict__ cnt3,
    const __bf16* __restrict__ Bsw,  // NT blocks of 4096 bf16
    const float* __restrict__ bias,  // 64
    int M) {
  const int lane = threadIdx.x & 31;
  const int wave = threadIdx.x >> 5;
  const int tn   = wave & 3;
  const int half = lane >> 4;
  const int l15  = lane & 15;
  const int col  = tn * 16 + l15;
  const float bcol = bias[col];

  const float* Aarr[4] = { A0, A1, A2, A3 };
  const float* Karr[4] = { cnt0, cnt1, cnt2, cnt3 };

  v16bf b[NT][2];
#pragma unroll
  for (int t = 0; t < NT; ++t)
#pragma unroll
    for (int kc = 0; kc < 2; ++kc)
      b[t][kc] = *(const v16bf*)(Bsw + (size_t)t * 4096 +
                                 ((size_t)(kc * 4 + tn) * 32 + lane) * 16);

  const int tm0 = blockIdx.x * (2 * TILES) + (wave >> 2) * TILES;
#pragma unroll
  for (int ti = 0; ti < TILES; ++ti) {
    const int tm = tm0 + ti;
    if (tm * 16 < M) {  // wave-uniform
      const int row = tm * 16 + l15;
      v8f c = {0.f, 0.f, 0.f, 0.f, 0.f, 0.f, 0.f, 0.f};
#pragma unroll
      for (int t = 0; t < NT; ++t) {
        const float* Ar = Aarr[t] + (size_t)row * Hdim;
        const float sc = Karr[t] ? (1.f / fmaxf(Karr[t][row], 1.f)) : 1.f;
#pragma unroll
        for (int kc = 0; kc < 2; ++kc) {
          const v16bf a = pack16s(Ar + kc * 32 + half * 8, sc);
          c = __builtin_amdgcn_wmma_f32_16x16x32_bf16(
              false, a, false, b[t][kc], (short)0, c, false, false);
        }
      }
      float* Crow = C + (size_t)(tm * 16) * Hdim + col;
#pragma unroll
      for (int r = 0; r < 8; ++r)
        Crow[(size_t)(r + 8 * half) * Hdim] = fmaxf(c[r] + bcol, 0.f);
    }
  }
}

// ---------------------------------------------------------------------------
// Edge message + scatter-sum:  sum[dst] += x_src[src] + ea@We + be ; cnt[dst]++
// 8 threads per edge, 8 features each; vectorized b128 reads, f32 atomics.
// ---------------------------------------------------------------------------
__global__ __launch_bounds__(256) void k_scatter_msg(
    const float* __restrict__ xsrc, const float* __restrict__ ea,
    const float* __restrict__ We, const float* __restrict__ be,
    const int* __restrict__ src, const int* __restrict__ dst, int E,
    float* __restrict__ sum, float* __restrict__ cnt) {
  const long long tid = (long long)blockIdx.x * blockDim.x + threadIdx.x;
  const long long e = tid >> 3;
  if (e >= E) return;
  const int chunk = (int)(tid & 7);
  const int base = chunk * 8;
  const int s = src[e];
  const int d = dst[e];
  const float4 av0 = *(const float4*)(ea + e * 8);
  const float4 av1 = *(const float4*)(ea + e * 8 + 4);
  const float av[8] = { av0.x, av0.y, av0.z, av0.w, av1.x, av1.y, av1.z, av1.w };
  const float4 x0 = *(const float4*)(xsrc + (size_t)s * Hdim + base);
  const float4 x1 = *(const float4*)(xsrc + (size_t)s * Hdim + base + 4);
  const float4 b0 = *(const float4*)(be + base);
  const float4 b1 = *(const float4*)(be + base + 4);
  float acc[8] = { x0.x + b0.x, x0.y + b0.y, x0.z + b0.z, x0.w + b0.w,
                   x1.x + b1.x, x1.y + b1.y, x1.z + b1.z, x1.w + b1.w };
#pragma unroll
  for (int k = 0; k < 8; ++k) {
    const float4 w0 = *(const float4*)(We + k * Hdim + base);
    const float4 w1 = *(const float4*)(We + k * Hdim + base + 4);
    acc[0] += av[k] * w0.x; acc[1] += av[k] * w0.y;
    acc[2] += av[k] * w0.z; acc[3] += av[k] * w0.w;
    acc[4] += av[k] * w1.x; acc[5] += av[k] * w1.y;
    acc[6] += av[k] * w1.z; acc[7] += av[k] * w1.w;
  }
  float* sr = sum + (size_t)d * Hdim + base;
#pragma unroll
  for (int c = 0; c < 8; ++c) atomicAdd(&sr[c], acc[c]);
  if (chunk == 0) atomicAdd(&cnt[d], 1.0f);
}

__global__ __launch_bounds__(256) void k_init_bias_emb(
    float* __restrict__ out, const float* __restrict__ b,
    const float* __restrict__ emb, int n64) {
  const int i = blockIdx.x * blockDim.x + threadIdx.x;
  if (i >= n64) return;
  out[i] = b[i & 63] + emb[i];
}

__global__ void k_sumBias(float* __restrict__ out, const float* __restrict__ b0,
                          const float* __restrict__ b1,
                          const float* __restrict__ b2) {
  const int i = threadIdx.x;  // 64 threads
  float v = b0[i];
  if (b1) v += b1[i];
  if (b2) v += b2[i];
  out[i] = v;
}

__global__ __launch_bounds__(256) void k_sumW3(
    float* __restrict__ out, const float* __restrict__ w0,
    const float* __restrict__ w1, const float* __restrict__ w2) {
  const int i = blockIdx.x * blockDim.x + threadIdx.x;
  if (i >= Hdim * Hdim) return;
  float v = w0[i];
  if (w1) v += w1[i];
  if (w2) v += w2[i];
  out[i] = v;
}

__global__ __launch_bounds__(256) void k_copy(float* __restrict__ out,
                                              const float* __restrict__ in,
                                              int n) {
  const int i = blockIdx.x * blockDim.x + threadIdx.x;
  if (i < n) out[i] = in[i];
}

// ---------------------------------------------------------------------------
// Classifier: one wave per 16 label edges (EL % 16 == 0).
// layer1: ef(128) @ W1(128x64) via 16 bf16 WMMAs (W1 pre-swizzled), relu->LDS
// layer2: h(64) @ W2(64x7) + b2 via VALU dot from LDS
// ---------------------------------------------------------------------------
__global__ __launch_bounds__(32) void k_classifier(
    const float* __restrict__ uh, const float* __restrict__ mh,
    const int* __restrict__ eu, const int* __restrict__ em,
    const __bf16* __restrict__ W1sw, const float* __restrict__ b1,
    const float* __restrict__ W2, const float* __restrict__ b2,
    float* __restrict__ out) {
  __shared__ float hbuf[16][68];  // stride 68 -> conflict-free layer-2 reads
  const int lane = threadIdx.x;
  const int tile = blockIdx.x;
  const int half = lane >> 4;
  const int l15  = lane & 15;
  const int erow = tile * 16 + l15;
  const float* urow = uh + (size_t)eu[erow] * Hdim;
  const float* mrow = mh + (size_t)em[erow] * Hdim;

  v8f acc[4];
#pragma unroll
  for (int t = 0; t < 4; ++t) acc[t] = (v8f){0.f,0.f,0.f,0.f,0.f,0.f,0.f,0.f};

#pragma unroll
  for (int kc = 0; kc < 4; ++kc) {  // K = 128: chunks 0,1 user / 2,3 movie
    const float* srcrow = (kc < 2) ? urow : mrow;
    const v16bf a = pack16(srcrow + (kc & 1) * 32 + half * 8);
#pragma unroll
    for (int tn = 0; tn < 4; ++tn) {
      const v16bf b =
          *(const v16bf*)(W1sw + ((size_t)(kc * 4 + tn) * 32 + lane) * 16);
      acc[tn] = __builtin_amdgcn_wmma_f32_16x16x32_bf16(false, a, false, b,
                                                        (short)0, acc[tn],
                                                        false, false);
    }
  }
#pragma unroll
  for (int tn = 0; tn < 4; ++tn) {
    const int colc = tn * 16 + l15;
#pragma unroll
    for (int r = 0; r < 8; ++r)
      hbuf[r + 8 * half][colc] = fmaxf(acc[tn][r] + b1[colc], 0.f);
  }
  __syncthreads();
  for (int idx = lane; idx < 112; idx += 32) {  // 16 edges x 7 outputs
    const int e = idx / 7, o = idx % 7;
    float s = b2[o];
#pragma unroll
    for (int k = 0; k < Hdim; ++k) s += hbuf[e][k] * W2[k * 7 + o];
    out[(size_t)(tile * 16 + e) * 7 + o] = s;
  }
}

// ---------------------------------------------------------------------------

static inline unsigned cdiv(long long a, long long b) {
  return (unsigned)((a + b - 1) / b);
}

extern "C" void kernel_launch(void* const* d_in, const int* in_sizes, int n_in,
                              void* d_out, int out_size, void* d_ws,
                              size_t ws_size, hipStream_t stream) {
  (void)in_sizes; (void)n_in; (void)out_size; (void)ws_size;

  const float* x_user       = (const float*)d_in[0];
  const float* x_movie      = (const float*)d_in[1];
  const float* user_emb     = (const float*)d_in[2];
  const float* movie_emb    = (const float*)d_in[3];
  const float* genre_emb    = (const float*)d_in[4];
  const float* director_emb = (const float*)d_in[5];
  const float* user_lin_w   = (const float*)d_in[6];
  const float* user_lin_b   = (const float*)d_in[7];
  const float* movie_lin_w  = (const float*)d_in[8];
  const float* movie_lin_b  = (const float*)d_in[9];
  // conv_params: ETYPES-order x L x {Wl,bl,Wr,We,be}, indices 10..69
  auto CP = [&](int et, int l, int f) -> const float* {
    return (const float*)d_in[10 + (et * 2 + l) * 5 + f];
  };
  const float* clf_w1 = (const float*)d_in[70];
  const float* clf_b1 = (const float*)d_in[71];
  const float* clf_w2 = (const float*)d_in[72];
  const float* clf_b2 = (const float*)d_in[73];
  const float* ea_rates = (const float*)d_in[74];
  const float* ea_hg    = (const float*)d_in[75];
  const float* ea_dir   = (const float*)d_in[76];
  const int* ei_rates = (const int*)d_in[77];
  const int* ei_hg    = (const int*)d_in[78];
  const int* ei_dir   = (const int*)d_in[79];
  const int* eli      = (const int*)d_in[80];

  // ---- workspace layout --------------------------------------------------
  float* w = (float*)d_ws;
  size_t off = 0;
  auto take = [&](size_t n) { float* p = w + off; off += n; return p; };
  float* bufA[4] = { take((size_t)NUc * Hdim), take((size_t)NMc * Hdim),
                     take((size_t)NGc * Hdim), take((size_t)NDc * Hdim) };
  float* bufB[4] = { take((size_t)NUc * Hdim), take((size_t)NMc * Hdim),
                     take((size_t)NGc * Hdim), take((size_t)NDc * Hdim) };
  float* agg[3] = { take((size_t)NUc * Hdim),   // covers any dst type
                    take((size_t)NMc * Hdim),   // movie terms 2,3
                    take((size_t)NMc * Hdim) };
  float* cntb[3] = { take((size_t)NUc), take((size_t)NMc), take((size_t)NMc) };
  float* wsum = take((size_t)Hdim * Hdim);
  float* bsum = take(64);
  __bf16* bswEnc = (__bf16*)take(2048);  // encode operand (<= 4096 bf16)
  __bf16* bswAll = (__bf16*)take(8192);  // up to 4 fused terms (16384 bf16)
  __bf16* w1sw   = (__bf16*)take(4096);  // classifier W1 (8192 bf16)

  float* cx[4] = { bufA[0], bufA[1], bufA[2], bufA[3] };
  float* co[4] = { bufB[0], bufB[1], bufB[2], bufB[3] };
  const int NSZ[4] = { NUc, NMc, NGc, NDc };

  const int TILES = 4;  // rows per block = 32*TILES = 128

  // ---- input encoders ----------------------------------------------------
  k_init_bias_emb<<<cdiv((long long)NUc * Hdim, 256), 256, 0, stream>>>(
      cx[0], user_lin_b, user_emb, NUc * Hdim);
  k_swizzleB<<<cdiv(32 * Hdim, 256), 256, 0, stream>>>(bswEnc, user_lin_w,
                                                       32 * Hdim);
  k_wmma_gemm_accum<32, 4><<<cdiv(NUc, 32 * TILES), 256, 0, stream>>>(
      cx[0], x_user, bswEnc, NUc);
  k_init_bias_emb<<<cdiv((long long)NMc * Hdim, 256), 256, 0, stream>>>(
      cx[1], movie_lin_b, movie_emb, NMc * Hdim);
  k_swizzleB<<<cdiv(64 * Hdim, 256), 256, 0, stream>>>(bswEnc, movie_lin_w,
                                                       64 * Hdim);
  k_wmma_gemm_accum<64, 4><<<cdiv(NMc, 32 * TILES), 256, 0, stream>>>(
      cx[1], x_movie, bswEnc, NMc);
  k_copy<<<cdiv(NGc * Hdim, 256), 256, 0, stream>>>(cx[2], genre_emb,
                                                    NGc * Hdim);
  k_copy<<<cdiv(NDc * Hdim, 256), 256, 0, stream>>>(cx[3], director_emb,
                                                    NDc * Hdim);

  // ---- edge specs (ETYPES order) ----------------------------------------
  struct ESpec { const int* src; const int* dst; int E; const float* ea; int st; int dt; };
  const ESpec es[6] = {
    { ei_rates,       ei_rates + E1c, E1c, ea_rates, 0, 1 },  // rates
    { ei_rates + E1c, ei_rates,       E1c, ea_rates, 1, 0 },  // rev_rates
    { ei_hg,          ei_hg + E2c,    E2c, ea_hg,    1, 2 },  // has_genre
    { ei_hg + E2c,    ei_hg,          E2c, ea_hg,    2, 1 },  // rev_has_genre
    { ei_dir,         ei_dir + E3c,   E3c, ea_dir,   1, 3 },  // directed
    { ei_dir + E3c,   ei_dir,         E3c, ea_dir,   3, 1 },  // rev_directed
  };

  // ---- conv layers -------------------------------------------------------
  for (int l = 0; l < 2; ++l) {
    for (int dt = 0; dt < 4; ++dt) {
      int ets[3], ne = 0;
      for (int e = 0; e < 6; ++e)
        if (es[e].dt == dt) ets[ne++] = e;
      const int M = NSZ[dt];
      const int n64 = M * Hdim;
      // bias = sum of incident bl's ; Wr_sum for the single x-pass
      k_sumBias<<<1, 64, 0, stream>>>(bsum, CP(ets[0], l, 1),
                                      ne > 1 ? CP(ets[1], l, 1) : nullptr,
                                      ne > 2 ? CP(ets[2], l, 1) : nullptr);
      k_sumW3<<<16, 256, 0, stream>>>(wsum, CP(ets[0], l, 2),
                                      ne > 1 ? CP(ets[1], l, 2) : nullptr,
                                      ne > 2 ? CP(ets[2], l, 2) : nullptr);
      k_swizzleB<<<cdiv(64 * Hdim, 256), 256, 0, stream>>>(bswAll, wsum,
                                                           64 * Hdim);
      // aggregate messages per incident edge type (raw sums + counts; the
      // mean divide is folded into the fused GEMM's A-operand scale)
      for (int i = 0; i < ne; ++i) {
        const ESpec& E = es[ets[i]];
        hipMemsetAsync(agg[i], 0, (size_t)n64 * sizeof(float), stream);
        hipMemsetAsync(cntb[i], 0, (size_t)M * sizeof(float), stream);
        k_scatter_msg<<<cdiv((long long)E.E * 8, 256), 256, 0, stream>>>(
            cx[E.st], E.ea, CP(ets[i], l, 3), CP(ets[i], l, 4), E.src, E.dst,
            E.E, agg[i], cntb[i]);
        k_swizzleB<<<cdiv(64 * Hdim, 256), 256, 0, stream>>>(
            bswAll + (size_t)(i + 1) * 4096, CP(ets[i], l, 0), 64 * Hdim);
      }
      // fused: out = relu(bias + x@sumWr + sum_i mean(agg_i)@Wl_i)
      const unsigned g = cdiv(M, 32 * TILES);
      if (ne == 1) {
        k_fused_out<2, 4><<<g, 256, 0, stream>>>(
            co[dt], cx[dt], nullptr, agg[0], cntb[0], nullptr, nullptr,
            nullptr, nullptr, bswAll, bsum, M);
      } else {
        k_fused_out<4, 4><<<g, 256, 0, stream>>>(
            co[dt], cx[dt], nullptr, agg[0], cntb[0], agg[1], cntb[1],
            agg[2], cntb[2], bswAll, bsum, M);
      }
    }
    for (int dt = 0; dt < 4; ++dt) { float* t = cx[dt]; cx[dt] = co[dt]; co[dt] = t; }
  }

  // ---- classifier --------------------------------------------------------
  k_swizzleB<<<cdiv(128 * Hdim, 256), 256, 0, stream>>>(w1sw, clf_w1,
                                                        128 * Hdim);
  k_classifier<<<ELc / 16, 32, 0, stream>>>(cx[0], cx[1], eli, eli + ELc,
                                            w1sw, clf_b1, clf_w2, clf_b2,
                                            (float*)d_out);
}
